// Decoder_25108378812749
// MI455X (gfx1250) — matrix operand order repaired and checked by reading, not compiled, not run on hardware
//
#include <hip/hip_runtime.h>
#include <hip/hip_bf16.h>

// ---------------- problem constants ----------------
#define BB   64      // batch
#define PP   196     // spatial positions (14*14)
#define EE   2048    // feature dim E
#define DD   256     // hidden D
#define HH   256     // attention hidden H
#define EMB  256     // embedding dim
#define VV   10000   // vocab
#define LL   32      // max caption length
#define TT   31      // time steps (L-1)
#define XK   2560    // EMB + EE + DD (fused [emb|gated ctx|h] input to gates GEMM)
#define GN   1024    // 4*D gate width

// ---------------- WMMA types (gfx1250, wave32) ----------------
typedef __attribute__((ext_vector_type(16))) __bf16 v16bf;
typedef __attribute__((ext_vector_type(8)))  float  v8f;

union Frag {        // one WMMA bf16 operand fragment per lane (8 VGPRs)
    v16bf  v;
    __bf16 e[16];
    uint4  q[2];
};

__device__ __forceinline__ __bf16 f2bf(float f) { return (__bf16)f; }  // native v_cvt, RNE

__device__ __forceinline__ float sigf(float x) { return 1.0f / (1.0f + __expf(-x)); }

__device__ __forceinline__ v8f wmma_bf16(const Frag& a, const Frag& b, v8f c) {
    return __builtin_amdgcn_wmma_f32_16x16x32_bf16(false, a.v, false, b.v,
                                                   (short)0, c, false, false);
}

// ---------------- 0: stable descending argsort of lengths ----------------
__global__ void sort_kernel(const int* __restrict__ len, int* __restrict__ order,
                            int* __restrict__ target) {
    int i = threadIdx.x;
    if (i < BB) {
        int li = len[i];
        int rank = 0;
        for (int j = 0; j < BB; ++j) {
            int lj = len[j];
            if (lj > li || (lj == li && j < i)) rank++;
        }
        order[rank]  = i;
        target[rank] = li - 1;
    }
}

// ---------------- 1: mean over spatial positions (sorted order) ----------------
__global__ void mean_kernel(const float* __restrict__ feat, const int* __restrict__ order,
                            float* __restrict__ mean_feat) {
    int b = blockIdx.x, tid = threadIdx.x;
    const float* fb = feat + (size_t)order[b] * PP * EE;
    for (int i = 0; i < 2; ++i) {
        int e = (tid + (i << 8)) << 2;                    // float4 per lane
        float4 s = make_float4(0.f, 0.f, 0.f, 0.f);
        for (int p = 0; p < PP; ++p) {
            float4 f = *(const float4*)(fb + (size_t)p * EE + e);
            s.x += f.x; s.y += f.y; s.z += f.z; s.w += f.w;
        }
        const float inv = 1.0f / (float)PP;
        s.x *= inv; s.y *= inv; s.z *= inv; s.w *= inv;
        *(float4*)(mean_feat + (size_t)b * EE + e) = s;
    }
}

// ---------------- 2: h0/c0 = mean_feat @ W_hid/W_cell + bias ----------------
__global__ void inithc_kernel(const float* __restrict__ mean_feat,
                              const float* __restrict__ W_hid, const float* __restrict__ b_hid,
                              const float* __restrict__ W_cell, const float* __restrict__ b_cell,
                              float* __restrict__ h, float* __restrict__ c) {
    int b = blockIdx.x, d = threadIdx.x;
    float ah = b_hid[d], ac = b_cell[d];
    const float* mf = mean_feat + (size_t)b * EE;
    for (int e = 0; e < EE; ++e) {
        float v = mf[e];
        ah += v * W_hid[(size_t)e * DD + d];
        ac += v * W_cell[(size_t)e * DD + d];
    }
    h[b * DD + d] = ah;
    c[b * DD + d] = ac;
}

// ---------------- 3: transpose+convert fp32 (KxN) -> bf16 (N x dstLD, at koff) --------
__global__ void transpose_bf16_kernel(const float* __restrict__ src, __bf16* __restrict__ dst,
                                      int K, int N, int dstLD, int dstKoff) {
    size_t idx = (size_t)blockIdx.x * blockDim.x + threadIdx.x;
    if (idx >= (size_t)K * N) return;
    int k = (int)(idx / N);
    int n = (int)(idx % N);
    dst[(size_t)n * dstLD + dstKoff + k] = f2bf(src[idx]);
}

// ---------------- 4: fe = feat[order] @ Wf + bf  (WMMA, fp32 A converted on the fly) ---
// grid 784, block 128 (4 waves); each wave: 16 rows x 64 cols (4 WMMA tiles share A frag)
__global__ void fe_gemm_kernel(const float* __restrict__ feat, const __bf16* __restrict__ WfT,
                               const float* __restrict__ bfv, const int* __restrict__ order,
                               float* __restrict__ fe) {
    const int K = EE, N = HH;
    int lane = threadIdx.x & 31;
    int wave = threadIdx.x >> 5;
    int row0 = blockIdx.x * 16;
    int colb = wave * 64;                        // wave covers cols [colb, colb+64)
    int m    = lane & 15;
    int r    = row0 + m;                         // logical (sorted) row 0..12543
    int b    = r / PP, p = r % PP;
    const float* arow = feat + ((size_t)order[b] * PP + p) * EE;
    int khiA = (lane & 16) ? 8 : 0;
    int khiB = (lane & 16) ? 8 : 0;
    const __bf16* brow = WfT + (size_t)(colb + m) * K + khiB;   // +16*K per col tile
    v8f acc0 = {}, acc1 = {}, acc2 = {}, acc3 = {};
    for (int k0 = 0; k0 < K; k0 += 32) {
        const float* ap = arow + k0 + khiA;
        if (k0 + 256 < K) __builtin_prefetch(ap + 256, 0, 0);   // stream feat from HBM
        float4 v0 = *(const float4*)(ap);
        float4 v1 = *(const float4*)(ap + 4);
        float4 v2 = *(const float4*)(ap + 16);
        float4 v3 = *(const float4*)(ap + 20);
        Frag a;
        a.e[0] = f2bf(v0.x); a.e[1] = f2bf(v0.y); a.e[2]  = f2bf(v0.z); a.e[3]  = f2bf(v0.w);
        a.e[4] = f2bf(v1.x); a.e[5] = f2bf(v1.y); a.e[6]  = f2bf(v1.z); a.e[7]  = f2bf(v1.w);
        a.e[8] = f2bf(v2.x); a.e[9] = f2bf(v2.y); a.e[10] = f2bf(v2.z); a.e[11] = f2bf(v2.w);
        a.e[12]= f2bf(v3.x); a.e[13]= f2bf(v3.y); a.e[14] = f2bf(v3.z); a.e[15] = f2bf(v3.w);
        const __bf16* bp = brow + k0;
        Frag b0, b1, b2, b3;
        b0.q[0] = *(const uint4*)(bp);                  b0.q[1] = *(const uint4*)(bp + 16);
        b1.q[0] = *(const uint4*)(bp + (size_t)16 * K); b1.q[1] = *(const uint4*)(bp + (size_t)16 * K + 16);
        b2.q[0] = *(const uint4*)(bp + (size_t)32 * K); b2.q[1] = *(const uint4*)(bp + (size_t)32 * K + 16);
        b3.q[0] = *(const uint4*)(bp + (size_t)48 * K); b3.q[1] = *(const uint4*)(bp + (size_t)48 * K + 16);
        acc0 = wmma_bf16(a, b0, acc0);
        acc1 = wmma_bf16(a, b1, acc1);
        acc2 = wmma_bf16(a, b2, acc2);
        acc3 = wmma_bf16(a, b3, acc3);
    }
    int mo = (lane & 16) ? 8 : 0;
    for (int j = 0; j < 4; ++j) {
        v8f acc = (j == 0) ? acc0 : (j == 1) ? acc1 : (j == 2) ? acc2 : acc3;
        int col = colb + j * 16 + m;
        float bias = bfv[col];
        for (int i = 0; i < 8; ++i)
            fe[(size_t)(row0 + mo + i) * N + col] = acc[i] + bias;
    }
}

// ---------------- 5: fused attention: he, scores, softmax, ctx, masked w out ----------
__global__ void att_kernel(const float* __restrict__ feat, const float* __restrict__ fe,
                           const float* __restrict__ h, const float* __restrict__ Wh,
                           const float* __restrict__ bh, const float* __restrict__ We,
                           const float* __restrict__ be, const int* __restrict__ order,
                           const int* __restrict__ target, int t,
                           float* __restrict__ att_out, float* __restrict__ ctx) {
    __shared__ float h_s[DD];
    __shared__ float he_s[HH];
    __shared__ float w_s[PP + 4];
    __shared__ float red[8];
    int b = blockIdx.x, tid = threadIdx.x;
    int lane = tid & 31, wave = tid >> 5;
    h_s[tid] = h[b * DD + tid];
    __syncthreads();
    { // he = h @ Wh + bh
        float acc = bh[tid];
        for (int k = 0; k < DD; ++k) acc += h_s[k] * Wh[k * HH + tid];
        he_s[tid] = acc;
    }
    __syncthreads();
    // scores e[p] = relu(fe[p]+he) . We + be   (one wave per position)
    const float* feb = fe + (size_t)b * PP * HH;
    for (int p = wave; p < PP; p += 8) {
        float acc = 0.f;
        const float* fr = feb + (size_t)p * HH;
        for (int k = lane; k < HH; k += 32) {
            float v = fr[k] + he_s[k];
            v = v > 0.f ? v : 0.f;
            acc += v * We[k];
        }
        for (int o = 16; o; o >>= 1) acc += __shfl_xor(acc, o, 32);
        if (lane == 0) w_s[p] = acc + be[0];
    }
    __syncthreads();
    // softmax over 196
    float v = (tid < PP) ? w_s[tid] : -3.0e38f;
    for (int o = 16; o; o >>= 1) v = fmaxf(v, __shfl_xor(v, o, 32));
    if (lane == 0) red[wave] = v;
    __syncthreads();
    float mx = red[0];
    for (int i = 1; i < 8; ++i) mx = fmaxf(mx, red[i]);
    float ex = (tid < PP) ? __expf(w_s[tid] - mx) : 0.f;
    float s = ex;
    for (int o = 16; o; o >>= 1) s += __shfl_xor(s, o, 32);
    __syncthreads();
    if (lane == 0) red[wave] = s;
    __syncthreads();
    float sm = 0.f;
    for (int i = 0; i < 8; ++i) sm += red[i];
    float wgt = ex / sm;
    if (tid < PP) {
        w_s[tid] = wgt;
        float msk = (target[b] > t) ? 1.f : 0.f;
        att_out[((size_t)b * TT + t) * PP + tid] = wgt * msk;
    }
    __syncthreads();
    // ctx = sum_p w[p] * feat[order[b]][p][:]   (float4 per lane, b128 loads)
    const float* fb = feat + (size_t)order[b] * PP * EE;
    for (int i = 0; i < 2; ++i) {
        int e = (tid + (i << 8)) << 2;
        float4 acc = make_float4(0.f, 0.f, 0.f, 0.f);
        for (int p = 0; p < PP; ++p) {
            float wv = w_s[p];
            float4 f = *(const float4*)(fb + (size_t)p * EE + e);
            acc.x += wv * f.x; acc.y += wv * f.y; acc.z += wv * f.z; acc.w += wv * f.w;
        }
        *(float4*)(ctx + (size_t)b * EE + e) = acc;
    }
}

// ---------------- 6: build x = [emb_t | sigmoid(h@W_gate+b)*ctx | h] in bf16 ----------
__global__ void build_x_kernel(const int* __restrict__ tok, const float* __restrict__ emb,
                               const float* __restrict__ h, const float* __restrict__ W_gate,
                               const float* __restrict__ b_gate, const float* __restrict__ ctx,
                               const int* __restrict__ order, int t, __bf16* __restrict__ x) {
    __shared__ float h_s[DD];
    int b = blockIdx.x, tid = threadIdx.x;
    h_s[tid] = h[b * DD + tid];
    __syncthreads();
    __bf16* xb = x + (size_t)b * XK;
    int token = tok[order[b] * LL + t];
    xb[tid]            = f2bf(emb[(size_t)token * EMB + tid]);
    xb[EMB + EE + tid] = f2bf(h_s[tid]);
    for (int i = 0; i < 2; ++i) {
        int e = (tid + (i << 8)) << 2;                   // float4 per lane
        float4 acc = *(const float4*)(b_gate + e);
        for (int k = 0; k < DD; ++k) {
            float hv = h_s[k];
            float4 wg = *(const float4*)(W_gate + (size_t)k * EE + e);
            acc.x += hv * wg.x; acc.y += hv * wg.y; acc.z += hv * wg.z; acc.w += hv * wg.w;
        }
        float4 cv = *(const float4*)(ctx + (size_t)b * EE + e);
        xb[EMB + e + 0] = f2bf(sigf(acc.x) * cv.x);
        xb[EMB + e + 1] = f2bf(sigf(acc.y) * cv.y);
        xb[EMB + e + 2] = f2bf(sigf(acc.z) * cv.z);
        xb[EMB + e + 3] = f2bf(sigf(acc.w) * cv.w);
    }
}

// ---------------- 7: gates GEMM  g = x(64x2560) @ WcatT^T  (WMMA bf16) ----------------
// grid (4, 4), block 128; each wave: 16 rows x 64 cols
__global__ void gates_gemm_kernel(const __bf16* __restrict__ A, const __bf16* __restrict__ Bt,
                                  float* __restrict__ C) {
    const int K = XK, N = GN;
    int lane = threadIdx.x & 31, wave = threadIdx.x >> 5;
    int row0 = blockIdx.x * 16;
    int colb = (blockIdx.y * 4 + wave) * 64;
    int m = lane & 15;
    int khiA = (lane & 16) ? 8 : 0;
    int khiB = (lane & 16) ? 8 : 0;
    const __bf16* ap   = A  + (size_t)(row0 + m) * K + khiA;
    const __bf16* brow = Bt + (size_t)(colb + m) * K + khiB;
    v8f acc0 = {}, acc1 = {}, acc2 = {}, acc3 = {};
    for (int k0 = 0; k0 < K; k0 += 32) {
        Frag a;
        a.q[0] = *(const uint4*)(ap + k0);
        a.q[1] = *(const uint4*)(ap + k0 + 16);
        const __bf16* bp = brow + k0;
        Frag b0, b1, b2, b3;
        b0.q[0] = *(const uint4*)(bp);                  b0.q[1] = *(const uint4*)(bp + 16);
        b1.q[0] = *(const uint4*)(bp + (size_t)16 * K); b1.q[1] = *(const uint4*)(bp + (size_t)16 * K + 16);
        b2.q[0] = *(const uint4*)(bp + (size_t)32 * K); b2.q[1] = *(const uint4*)(bp + (size_t)32 * K + 16);
        b3.q[0] = *(const uint4*)(bp + (size_t)48 * K); b3.q[1] = *(const uint4*)(bp + (size_t)48 * K + 16);
        acc0 = wmma_bf16(a, b0, acc0);
        acc1 = wmma_bf16(a, b1, acc1);
        acc2 = wmma_bf16(a, b2, acc2);
        acc3 = wmma_bf16(a, b3, acc3);
    }
    int mo = (lane & 16) ? 8 : 0;
    for (int j = 0; j < 4; ++j) {
        v8f acc = (j == 0) ? acc0 : (j == 1) ? acc1 : (j == 2) ? acc2 : acc3;
        int col = colb + j * 16 + m;
        for (int i = 0; i < 8; ++i)
            C[(size_t)(row0 + mo + i) * N + col] = acc[i];
    }
}

// ---------------- 8: LSTM elementwise update ----------------
__global__ void lstm_kernel(const float* __restrict__ g, const float* __restrict__ b_ih,
                            const float* __restrict__ b_hh, const int* __restrict__ target,
                            int t, float* __restrict__ h, float* __restrict__ c,
                            __bf16* __restrict__ hnew_bf) {
    int b = blockIdx.x, d = threadIdx.x;
    const float* gb = g + (size_t)b * GN;
    float gi = gb[d]           + b_ih[d]           + b_hh[d];
    float gf = gb[DD + d]      + b_ih[DD + d]      + b_hh[DD + d];
    float gg = gb[2 * DD + d]  + b_ih[2 * DD + d]  + b_hh[2 * DD + d];
    float go = gb[3 * DD + d]  + b_ih[3 * DD + d]  + b_hh[3 * DD + d];
    float cold = c[b * DD + d];
    float cn = sigf(gf) * cold + sigf(gi) * tanhf(gg);
    float hn = sigf(go) * tanhf(cn);
    hnew_bf[b * DD + d] = f2bf(hn);
    if (target[b] > t) {
        h[b * DD + d] = hn;
        c[b * DD + d] = cn;
    }
}

// ---------------- 9: head GEMM  pred = h_new(64x256) @ W_head, masked (WMMA bf16) -----
// grid (4, 40), block 128; each wave: 16 rows x 4 N-tiles (625 tiles total, clamped)
__global__ void head_gemm_kernel(const __bf16* __restrict__ A, const __bf16* __restrict__ Bt,
                                 const float* __restrict__ bias, const int* __restrict__ target,
                                 int t, float* __restrict__ out) {
    const int K = DD, NT = VV / 16;              // 625 column tiles
    int lane = threadIdx.x & 31, wave = threadIdx.x >> 5;
    int chunk = blockIdx.y * 4 + wave;           // 0..159
    int ntb = chunk * 4;
    if (ntb >= NT) return;
    int row0 = blockIdx.x * 16;
    int m = lane & 15;
    int khiA = (lane & 16) ? 8 : 0;
    int khiB = (lane & 16) ? 8 : 0;
    int nt0 = ntb, nt1 = min(ntb + 1, NT - 1), nt2 = min(ntb + 2, NT - 1), nt3 = min(ntb + 3, NT - 1);
    const __bf16* ap  = A  + (size_t)(row0 + m) * K + khiA;
    const __bf16* bp0 = Bt + (size_t)(nt0 * 16 + m) * K + khiB;
    const __bf16* bp1 = Bt + (size_t)(nt1 * 16 + m) * K + khiB;
    const __bf16* bp2 = Bt + (size_t)(nt2 * 16 + m) * K + khiB;
    const __bf16* bp3 = Bt + (size_t)(nt3 * 16 + m) * K + khiB;
    v8f acc0 = {}, acc1 = {}, acc2 = {}, acc3 = {};
    for (int k0 = 0; k0 < K; k0 += 32) {
        Frag a, b0, b1, b2, b3;
        a.q[0]  = *(const uint4*)(ap + k0);   a.q[1]  = *(const uint4*)(ap + k0 + 16);
        b0.q[0] = *(const uint4*)(bp0 + k0);  b0.q[1] = *(const uint4*)(bp0 + k0 + 16);
        b1.q[0] = *(const uint4*)(bp1 + k0);  b1.q[1] = *(const uint4*)(bp1 + k0 + 16);
        b2.q[0] = *(const uint4*)(bp2 + k0);  b2.q[1] = *(const uint4*)(bp2 + k0 + 16);
        b3.q[0] = *(const uint4*)(bp3 + k0);  b3.q[1] = *(const uint4*)(bp3 + k0 + 16);
        acc0 = wmma_bf16(a, b0, acc0);
        acc1 = wmma_bf16(a, b1, acc1);
        acc2 = wmma_bf16(a, b2, acc2);
        acc3 = wmma_bf16(a, b3, acc3);
    }
    int mo = (lane & 16) ? 8 : 0;
    for (int j = 0; j < 4; ++j) {
        if (ntb + j >= NT) break;
        v8f acc = (j == 0) ? acc0 : (j == 1) ? acc1 : (j == 2) ? acc2 : acc3;
        int col = (ntb + j) * 16 + m;
        float bn = bias[col];
        for (int i = 0; i < 8; ++i) {
            int rb = row0 + mo + i;
            float msk = (target[rb] > t) ? 1.f : 0.f;
            out[((size_t)rb * TT + t) * VV + col] = msk * (acc[i] + bn);
        }
    }
}

// ---------------- launcher ----------------
extern "C" void kernel_launch(void* const* d_in, const int* in_sizes, int n_in,
                              void* d_out, int out_size, void* d_ws, size_t ws_size,
                              hipStream_t stream) {
    const float* image_feature = (const float*)d_in[0];
    const int*   caption_token = (const int*)d_in[1];
    const int*   caption_length= (const int*)d_in[2];
    const float* Wf     = (const float*)d_in[3];
    const float* bfv    = (const float*)d_in[4];
    const float* Wh     = (const float*)d_in[5];
    const float* bh     = (const float*)d_in[6];
    const float* We     = (const float*)d_in[7];
    const float* be     = (const float*)d_in[8];
    const float* emb    = (const float*)d_in[9];
    const float* W_ih   = (const float*)d_in[10];
    const float* b_ih   = (const float*)d_in[11];
    const float* W_hh   = (const float*)d_in[12];
    const float* b_hh   = (const float*)d_in[13];
    const float* W_hid  = (const float*)d_in[14];
    const float* b_hid  = (const float*)d_in[15];
    const float* W_cell = (const float*)d_in[16];
    const float* b_cell = (const float*)d_in[17];
    const float* W_gate = (const float*)d_in[18];
    const float* b_gate = (const float*)d_in[19];
    const float* W_head = (const float*)d_in[20];
    const float* b_head = (const float*)d_in[21];

    float* out_pred = (float*)d_out;                       // (B, T, V)
    float* out_att  = out_pred + (size_t)BB * TT * VV;     // (B, T, 196)

    // workspace carve-out (256B aligned)
    char* w = (char*)d_ws;
    auto alloc = [&](size_t bytes) { char* p = w; w += (bytes + 255) & ~(size_t)255; return p; };
    int*    order     = (int*)alloc(BB * 4);
    int*    target    = (int*)alloc(BB * 4);
    float*  mean_feat = (float*)alloc((size_t)BB * EE * 4);
    float*  h         = (float*)alloc((size_t)BB * DD * 4);
    float*  c         = (float*)alloc((size_t)BB * DD * 4);
    __bf16* hnew      = (__bf16*)alloc((size_t)BB * DD * 2);
    float*  ctx       = (float*)alloc((size_t)BB * EE * 4);
    __bf16* xcat      = (__bf16*)alloc((size_t)BB * XK * 2);
    float*  gbuf      = (float*)alloc((size_t)BB * GN * 4);
    float*  fe        = (float*)alloc((size_t)BB * PP * HH * 4);
    __bf16* WfT       = (__bf16*)alloc((size_t)HH * EE * 2);
    __bf16* WcatT     = (__bf16*)alloc((size_t)GN * XK * 2);
    __bf16* WheadT    = (__bf16*)alloc((size_t)VV * DD * 2);

    // one-time setup
    sort_kernel<<<1, 64, 0, stream>>>(caption_length, order, target);
    mean_kernel<<<BB, 256, 0, stream>>>(image_feature, order, mean_feat);
    inithc_kernel<<<BB, 256, 0, stream>>>(mean_feat, W_hid, b_hid, W_cell, b_cell, h, c);

    transpose_bf16_kernel<<<(EE * HH + 255) / 256, 256, 0, stream>>>(Wf, WfT, EE, HH, EE, 0);
    transpose_bf16_kernel<<<((EMB + EE) * GN + 255) / 256, 256, 0, stream>>>(W_ih, WcatT, EMB + EE, GN, XK, 0);
    transpose_bf16_kernel<<<(DD * GN + 255) / 256, 256, 0, stream>>>(W_hh, WcatT, DD, GN, XK, EMB + EE);
    transpose_bf16_kernel<<<(DD * VV + 255) / 256, 256, 0, stream>>>(W_head, WheadT, DD, VV, DD, 0);

    fe_gemm_kernel<<<BB * PP / 16, 128, 0, stream>>>(image_feature, WfT, bfv, order, fe);

    // recurrent steps
    for (int t = 0; t < TT; ++t) {
        att_kernel<<<BB, 256, 0, stream>>>(image_feature, fe, h, Wh, bh, We, be,
                                           order, target, t, out_att, ctx);
        build_x_kernel<<<BB, 256, 0, stream>>>(caption_token, emb, h, W_gate, b_gate,
                                               ctx, order, t, xcat);
        gates_gemm_kernel<<<dim3(4, 4), 128, 0, stream>>>(xcat, WcatT, gbuf);
        lstm_kernel<<<BB, 256, 0, stream>>>(gbuf, b_ih, b_hh, target, t, h, c, hnew);
        head_gemm_kernel<<<dim3(4, 40), 128, 0, stream>>>(hnew, WheadT, b_head, target, t, out_pred);
    }
    (void)in_sizes; (void)n_in; (void)out_size; (void)ws_size;
}
